// NystromAttention_47339129536647
// MI455X (gfx1250) — matrix operand order repaired and checked by reading, not compiled, not run on hardware
//
#include <hip/hip_runtime.h>
#include <cstdint>
#include <cstddef>

// ---------------------------------------------------------------- constants
#define BB 4
#define SS 4096
#define DD 512
#define HH 8
#define HD 64
#define LL 256
#define NBH 32          // B*H
#define QSCALE 0.125f   // hd^-0.5 = 64^-0.5

typedef __attribute__((ext_vector_type(16))) _Float16 v16h;
typedef __attribute__((ext_vector_type(8)))  float    v8f;

// ---------------------------------------------------------------- WMMA core
__device__ inline v8f wmma32(v16h a, v16h b, v8f c) {
  // D = A(16x32 f16) x B(32x16 f16) + C(16x16 f32)
  return __builtin_amdgcn_wmma_f32_16x16x32_f16(
      /*neg_a=*/false, a, /*neg_b=*/false, b,
      /*c_mod=*/(short)0, c, /*reuse_a=*/false, /*reuse_b=*/false);
}

// A fragment: 16x32 f16 (ISA 7.12.2): lane m=lane&15, half=lane>>4,
// VGPR r<4 -> K = half*8 + 2r ; r>=4 -> K = 16 + half*8 + 2(r-4)
__device__ inline v16h frag_a(const _Float16* base, int rowStride, int lane, int kofs) {
  int m = lane & 15, hf = lane >> 4;
  v16h f;
#pragma unroll
  for (int r = 0; r < 8; ++r) {
    int kb = (r < 4) ? (hf * 8 + 2 * r) : (16 + hf * 8 + 2 * (r - 4));
    kb += kofs;
    const _Float16* p = base + (size_t)m * rowStride + kb;
    f[2 * r]     = p[0];
    f[2 * r + 1] = p[1];
  }
  return f;
}

// B fragment: 32x16 f16: lane n=lane&15, half=lane>>4, VGPR r<4 -> K = half*8 + 2r ; r>=4 -> K = 16 + half*8 + 2(r-4)
// base is "column-major" staging: base[n*rowStride + k] == B[k][n]
__device__ inline v16h frag_b(const _Float16* base, int rowStride, int lane, int kofs) {
  int n = lane & 15, hf = lane >> 4;
  v16h f;
#pragma unroll
  for (int r = 0; r < 8; ++r) {
    int kb = ((r < 4) ? (hf * 8 + 2 * r) : (16 + hf * 8 + 2 * (r - 4))) + kofs;
    const _Float16* p = base + (size_t)n * rowStride + kb;
    f[2 * r]     = p[0];
    f[2 * r + 1] = p[1];
  }
  return f;
}

// ---------------------------------------------------------------- kernel 1:
// qkv = x @ Wqkv, scatter into q(f16, scaled) k(f16) [bh,s,64]; v^T(f16) [bh,64,s]
__global__ __launch_bounds__(128) void gemm_qkv_kernel(
    const float* __restrict__ X, const float* __restrict__ W,
    _Float16* __restrict__ qh, _Float16* __restrict__ kh, _Float16* __restrict__ vt) {
  const int N = 3 * DD, K = DD;
  __shared__ _Float16 Al[64 * 32];   // [row][k]
  __shared__ _Float16 Bl[64 * 32];   // [n][k]
  int m0 = blockIdx.x * 64, n0 = blockIdx.y * 64;
  int tid = threadIdx.x, lane = tid & 31, w = tid >> 5;
  v8f acc[4] = {};
  for (int k0 = 0; k0 < K; k0 += 32) {
    {
      int row = tid >> 1, c0 = (tid & 1) * 16;
      const float* s = X + (size_t)(m0 + row) * K + k0 + c0;
      _Float16* d = Al + row * 32 + c0;
#pragma unroll
      for (int i = 0; i < 16; ++i) d[i] = (_Float16)s[i];
      if (k0 + 32 < K) __builtin_prefetch(s + 32, 0, 1);  // global_prefetch_b8
    }
    {
      int n = tid >> 1, c0 = (tid & 1) * 16;
      _Float16* d = Bl + n * 32 + c0;
#pragma unroll
      for (int i = 0; i < 16; ++i)
        d[i] = (_Float16)W[(size_t)(k0 + c0 + i) * N + (n0 + n)];
    }
    __syncthreads();
    v16h af = frag_a(Al + w * 16 * 32, 32, lane, 0);
#pragma unroll
    for (int nt = 0; nt < 4; ++nt) {
      v16h bf = frag_b(Bl + nt * 16 * 32, 32, lane, 0);
      acc[nt] = wmma32(af, bf, acc[nt]);
    }
    __syncthreads();
  }
  int hf = lane >> 4, nn = lane & 15;
#pragma unroll
  for (int nt = 0; nt < 4; ++nt)
#pragma unroll
    for (int r = 0; r < 8; ++r) {
      int mg = m0 + w * 16 + hf * 8 + r;
      int ng = n0 + nt * 16 + nn;
      float v = acc[nt][r];
      int b = mg >> 12, s = mg & (SS - 1);
      int which = ng / DD, d = ng - which * DD;
      int h = d >> 6, e = d & 63;
      size_t bh = (size_t)(b * HH + h);
      if (which == 0)      qh[(bh * SS + s) * HD + e] = (_Float16)(v * QSCALE);
      else if (which == 1) kh[(bh * SS + s) * HD + e] = (_Float16)v;
      else                 vt[(bh * HD + e) * SS + s] = (_Float16)v;
    }
}

// ---------------------------------------------------------------- kernel 2:
// landmark means: q_lm/k_lm = mean over 16-token groups -> f32 and f16 copies
__global__ __launch_bounds__(64) void landmarks_kernel(
    const _Float16* __restrict__ qh, const _Float16* __restrict__ kh,
    float* __restrict__ qlm32, float* __restrict__ klm32,
    _Float16* __restrict__ qlm16, _Float16* __restrict__ klm16) {
  int e = threadIdx.x, l = blockIdx.x, bh = blockIdx.y, sel = blockIdx.z;
  const _Float16* src = sel ? kh : qh;
  float sum = 0.f;
#pragma unroll
  for (int j = 0; j < 16; ++j)
    sum += (float)src[((size_t)bh * SS + l * 16 + j) * HD + e];
  sum *= (1.0f / 16.0f);
  size_t idx = ((size_t)bh * LL + l) * HD + e;
  if (sel) { klm32[idx] = sum; klm16[idx] = (_Float16)sum; }
  else     { qlm32[idx] = sum; qlm16[idx] = (_Float16)sum; }
}

// ---------------------------------------------------------------- kernel 3:
// generic batched WMMA GEMM: out = A[batch,M,K] @ B[batch,K,N] (or B[batch,N,K] if bTrans)
// epilogues: outPlain = acc ; outAff = affA*acc + affD*I ; outHT[ng*M+mg] = f16(affA*acc)
__global__ __launch_bounds__(128) void gemm_batched_kernel(
    const float* __restrict__ A, const float* __restrict__ Bsrc,
    int M, int N, int K, int bTrans,
    float* __restrict__ outPlain, float* __restrict__ outAff,
    float affA, float affD, _Float16* __restrict__ outHT) {
  __shared__ _Float16 Al[64 * 32];   // [row][k]
  __shared__ _Float16 Bl[64 * 32];   // [n][k]
  int batch = blockIdx.z;
  int m0 = blockIdx.x * 64, n0 = blockIdx.y * 64;
  const float* Ab = A + (size_t)batch * M * K;
  const float* Bb = Bsrc + (size_t)batch * K * N;
  int tid = threadIdx.x, lane = tid & 31, w = tid >> 5;
  v8f acc[4] = {};
  for (int k0 = 0; k0 < K; k0 += 32) {
    {
      int row = tid >> 1, c0 = (tid & 1) * 16;
      const float* s = Ab + (size_t)(m0 + row) * K + k0 + c0;
      _Float16* d = Al + row * 32 + c0;
#pragma unroll
      for (int i = 0; i < 16; ++i) d[i] = (_Float16)s[i];
      if (k0 + 32 < K) __builtin_prefetch(s + 32, 0, 1);
    }
    {
      int n = tid >> 1, c0 = (tid & 1) * 16;
      _Float16* d = Bl + n * 32 + c0;
      if (bTrans) {
        const float* s = Bb + (size_t)(n0 + n) * K + k0 + c0;
#pragma unroll
        for (int i = 0; i < 16; ++i) d[i] = (_Float16)s[i];
      } else {
#pragma unroll
        for (int i = 0; i < 16; ++i)
          d[i] = (_Float16)Bb[(size_t)(k0 + c0 + i) * N + (n0 + n)];
      }
    }
    __syncthreads();
    v16h af = frag_a(Al + w * 16 * 32, 32, lane, 0);
#pragma unroll
    for (int nt = 0; nt < 4; ++nt) {
      v16h bf = frag_b(Bl + nt * 16 * 32, 32, lane, 0);
      acc[nt] = wmma32(af, bf, acc[nt]);
    }
    __syncthreads();
  }
  int hf = lane >> 4, nn = lane & 15;
#pragma unroll
  for (int nt = 0; nt < 4; ++nt)
#pragma unroll
    for (int r = 0; r < 8; ++r) {
      int mg = m0 + w * 16 + hf * 8 + r;
      int ng = n0 + nt * 16 + nn;
      float v = acc[nt][r];
      size_t idx = (size_t)batch * M * N + (size_t)mg * N + ng;
      if (outPlain) outPlain[idx] = v;
      if (outAff)   outAff[idx] = affA * v + (mg == ng ? affD : 0.0f);
      if (outHT)    outHT[(size_t)batch * M * N + (size_t)ng * M + mg] = (_Float16)(affA * v);
    }
}

// ---------------------------------------------------------------- softmax rows (len 256)
__global__ __launch_bounds__(256) void softmax_rows_kernel(float* __restrict__ x) {
  __shared__ float red[256];
  float* p = x + (size_t)blockIdx.x * LL;
  int t = threadIdx.x;
  float v = p[t];
  red[t] = v; __syncthreads();
  for (int s = 128; s > 0; s >>= 1) { if (t < s) red[t] = fmaxf(red[t], red[t + s]); __syncthreads(); }
  float mx = red[0]; __syncthreads();
  float e = __expf(v - mx);
  red[t] = e; __syncthreads();
  for (int s = 128; s > 0; s >>= 1) { if (t < s) red[t] += red[t + s]; __syncthreads(); }
  p[t] = e / red[0];
}

// ---------------------------------------------------------------- pinv init scalar
__global__ void init_scalar_kernel(unsigned* p) { *p = 0u; }

// column sums of attn2 (positive), global max via uint atomicMax (order-preserving for +floats)
__global__ __launch_bounds__(256) void colsum_max_kernel(const float* __restrict__ x,
                                                         unsigned* __restrict__ bits) {
  int bh = blockIdx.x, c = threadIdx.x;
  const float* p = x + (size_t)bh * LL * LL;
  float s = 0.f;
  for (int r = 0; r < LL; ++r) s += p[(size_t)r * LL + c];
  atomicMax(bits, __float_as_uint(s));
}

// z0 = attn2^T / maxColSum   (row-sums of softmax are exactly 1 -> max(col)==1)
__global__ __launch_bounds__(256) void init_z_kernel(const float* __restrict__ x,
                                                     const unsigned* __restrict__ bits,
                                                     float* __restrict__ z) {
  int i = blockIdx.x, bh = blockIdx.y, j = threadIdx.x;
  float s = __uint_as_float(*bits);
  z[((size_t)bh * LL + i) * LL + j] = x[((size_t)bh * LL + j) * LL + i] / s;
}

// ---------------------------------------------------------------- fused flash attention:
// out = softmax(Q @ K^T) @ V ;  Q [bh,nq,64] f16 (pre-scaled), K [bh,nk,64] f16,
// V given transposed: Vt [bh,64,nk] f16.  Online softmax over nk in chunks of 64.
// Q tile is filled with GLOBAL_LOAD_ASYNC_TO_LDS_B128 (ASYNCcnt path).
__global__ __launch_bounds__(128) void fused_attn_kernel(
    const _Float16* __restrict__ Q, const _Float16* __restrict__ Kh,
    const _Float16* __restrict__ Vt, int nq, int nk,
    float* __restrict__ out, size_t sB, size_t sH, size_t sRow) {
  __shared__ _Float16 Qs[16 * 64];
  __shared__ float    Sc[16 * 64];
  __shared__ _Float16 Pl[16 * 64];
  __shared__ float    Ac[16 * 64];
  __shared__ float    red[128];
  __shared__ float    mrun[16], lrun[16], corr[16];
  int bh = blockIdx.y, q0 = blockIdx.x * 16;
  int tid = threadIdx.x, lane = tid & 31, w = tid >> 5;
  int hf = lane >> 4, nn = lane & 15;
  int rowt = tid >> 3, g = tid & 7;   // 16 rows x 8 groups for softmax phases

  // async LDS fill of the 2KB Q tile: 128 lanes x 16B each, no conversion needed
  const _Float16* Qb = Q + ((size_t)bh * nq + q0) * HD;
  {
    unsigned ldsa = (unsigned)(uintptr_t)&Qs[tid * 8];
    const _Float16* ga = Qb + tid * 8;
    asm volatile("global_load_async_to_lds_b128 %0, %1, off"
                 :: "v"(ldsa), "v"(ga) : "memory");
  }
  for (int i = tid; i < 16 * 64; i += 128) Ac[i] = 0.f;
  if (tid < 16) { mrun[tid] = -1e30f; lrun[tid] = 0.f; }
  asm volatile("s_wait_asynccnt 0" ::: "memory");
  __syncthreads();

  const _Float16* Kb = Kh + (size_t)bh * nk * HD;
  const _Float16* Vb = Vt + (size_t)bh * HD * nk;
  int nchunks = nk / 64;
  for (int ch = 0; ch < nchunks; ++ch) {
    // ---- scores: each wave -> 16q x 16keys tile, K-dim = 64 (2 wmma steps)
    int keyBase = ch * 64 + w * 16;
    v8f s = {};
#pragma unroll
    for (int ks = 0; ks < 2; ++ks) {
      v16h af = frag_a(Qs, 64, lane, ks * 32);
      v16h bf = frag_b(Kb + (size_t)keyBase * HD, HD, lane, ks * 32);  // B[k][j]=K[j][k]
      s = wmma32(af, bf, s);
    }
#pragma unroll
    for (int r = 0; r < 8; ++r)
      Sc[(hf * 8 + r) * 64 + w * 16 + nn] = s[r];
    __syncthreads();                                   // (1) scores visible
    // ---- online softmax: 8 threads per row
    {
      const float* sr = Sc + rowt * 64 + g * 8;
      float mc = -1e30f;
#pragma unroll
      for (int c = 0; c < 8; ++c) mc = fmaxf(mc, sr[c]);
      red[tid] = mc;
    }
    __syncthreads();                                   // (2) partial maxes
    if (g == 0) {
      float mc = red[rowt * 8];
#pragma unroll
      for (int j = 1; j < 8; ++j) mc = fmaxf(mc, red[rowt * 8 + j]);
      float mo = mrun[rowt], mn = fmaxf(mo, mc);
      corr[rowt] = __expf(mo - mn);
      mrun[rowt] = mn;
    }
    __syncthreads();                                   // (3) mrun/corr ready
    {
      float mn = mrun[rowt];
      float sum = 0.f;
#pragma unroll
      for (int c = 0; c < 8; ++c) {
        int idx = rowt * 64 + g * 8 + c;
        float e = __expf(Sc[idx] - mn);
        sum += e;
        Pl[idx] = (_Float16)e;
      }
      red[tid] = sum;
    }
    __syncthreads();                                   // (4) Pl + partial sums
    if (g == 0) {
      float sum = 0.f;
#pragma unroll
      for (int j = 0; j < 8; ++j) sum += red[rowt * 8 + j];
      lrun[rowt] = lrun[rowt] * corr[rowt] + sum;      // owned per-thread, no race
    }
    // ---- P(16x64) @ Vchunk(64x64): wave w -> out cols [16w,16w+16)
    v8f o = {};
#pragma unroll
    for (int ks = 0; ks < 2; ++ks) {
      v16h af = frag_a(Pl, 64, lane, ks * 32);
      v16h bf = frag_b(Vb + (size_t)(w * 16) * nk + ch * 64, nk, lane, ks * 32);
      o = wmma32(af, bf, o);
    }
#pragma unroll
    for (int r = 0; r < 8; ++r) {
      int m = hf * 8 + r, e = w * 16 + nn;
      Ac[m * 64 + e] = Ac[m * 64 + e] * corr[m] + o[r];
    }
    __syncthreads();                                   // (5) end of chunk
  }
  // ---- writeout with flexible strides (head-merge or [bh,l,e])
  int b = bh >> 3, h = bh & 7;
  float* ob = out + (size_t)b * sB + (size_t)h * sH;
  for (int i = tid; i < 16 * 64; i += 128) {
    int m = i >> 6, e = i & 63;
    ob[(size_t)(q0 + m) * sRow + e] = Ac[i] / lrun[m];
  }
}

// ---------------------------------------------------------------- launcher
extern "C" void kernel_launch(void* const* d_in, const int* in_sizes, int n_in,
                              void* d_out, int out_size, void* d_ws, size_t ws_size,
                              hipStream_t stream) {
  (void)in_sizes; (void)n_in; (void)out_size; (void)ws_size;
  const float* x    = (const float*)d_in[0];   // [4,4096,512]
  const float* Wqkv = (const float*)d_in[1];   // [512,1536]
  float* out = (float*)d_out;                  // [4,4096,512]

  char* ws = (char*)d_ws;
  size_t off = 0;
  auto alloc = [&](size_t bytes) -> void* {
    off = (off + 255) & ~(size_t)255;
    void* p = ws + off;
    off += bytes;
    return p;
  };
  const size_t shS = (size_t)NBH * SS * HD;   // per-tensor seq elems
  const size_t shL = (size_t)NBH * LL * HD;
  const size_t mm  = (size_t)NBH * LL * LL;

  _Float16* qh    = (_Float16*)alloc(shS * 2);
  _Float16* kh    = (_Float16*)alloc(shS * 2);
  _Float16* vt    = (_Float16*)alloc(shS * 2);
  float*    qlm32 = (float*)   alloc(shL * 4);
  float*    klm32 = (float*)   alloc(shL * 4);
  _Float16* qlm16 = (_Float16*)alloc(shL * 2);
  _Float16* klm16 = (_Float16*)alloc(shL * 2);
  float*    xbuf  = (float*)   alloc(mm * 4);  // attn2
  float*    zA    = (float*)   alloc(mm * 4);
  float*    zB    = (float*)   alloc(mm * 4);
  float*    xz    = (float*)   alloc(mm * 4);
  float*    t1    = (float*)   alloc(mm * 4);
  float*    t2    = (float*)   alloc(mm * 4);
  float*    w1    = (float*)   alloc(shL * 4); // attn3 @ v : [bh,256,64]
  _Float16* w2t   = (_Float16*)alloc(shL * 2); // (z @ w1)^T : [bh,64,256]
  unsigned* bits  = (unsigned*)alloc(256);

  // 1) QKV projection -> q/k f16 [bh,s,64], v^T f16 [bh,64,s]
  gemm_qkv_kernel<<<dim3((BB * SS) / 64, (3 * DD) / 64, 1), 128, 0, stream>>>(
      x, Wqkv, qh, kh, vt);

  // 2) landmarks
  landmarks_kernel<<<dim3(LL, NBH, 2), 64, 0, stream>>>(qh, kh, qlm32, klm32, qlm16, klm16);

  // 3) sim2 = q_lm @ k_lm^T  (B transposed), then row softmax -> attn2
  gemm_batched_kernel<<<dim3(LL / 64, LL / 64, NBH), 128, 0, stream>>>(
      qlm32, klm32, LL, LL, HD, /*bTrans=*/1, xbuf, nullptr, 0.f, 0.f, nullptr);
  softmax_rows_kernel<<<dim3(NBH * LL), 256, 0, stream>>>(xbuf);

  // 4) pinv init: z0 = attn2^T / max(colsum)   (rowsums == 1)
  init_scalar_kernel<<<1, 1, 0, stream>>>(bits);
  colsum_max_kernel<<<dim3(NBH), 256, 0, stream>>>(xbuf, bits);
  init_z_kernel<<<dim3(LL, NBH), 256, 0, stream>>>(xbuf, bits, zA);

  // 5) Newton-Schulz: z = 0.25 z (13I - xz(15I - xz(7I - xz)))
  float* zc = zA;
  float* zn = zB;
  dim3 gP(LL / 64, LL / 64, NBH);
  for (int it = 0; it < 6; ++it) {
    gemm_batched_kernel<<<gP, 128, 0, stream>>>(xbuf, zc, LL, LL, LL, 0,
                                                xz, t1, -1.f, 7.f, nullptr);   // xz; t1 = 7I - xz
    gemm_batched_kernel<<<gP, 128, 0, stream>>>(xz, t1, LL, LL, LL, 0,
                                                nullptr, t2, -1.f, 15.f, nullptr); // t2 = 15I - xz@t1
    gemm_batched_kernel<<<gP, 128, 0, stream>>>(xz, t2, LL, LL, LL, 0,
                                                nullptr, t1, -1.f, 13.f, nullptr); // t1 = 13I - xz@t2
    gemm_batched_kernel<<<gP, 128, 0, stream>>>(zc, t1, LL, LL, LL, 0,
                                                nullptr, zn, 0.25f, 0.f, nullptr); // zn = 0.25 z@t1
    float* tmp = zc; zc = zn; zn = tmp;
  }

  // 6) w1 = softmax(q_lm @ k^T) @ v    (fused flash over S=4096) -> fp32 [bh,256,64]
  fused_attn_kernel<<<dim3(LL / 16, NBH), 128, 0, stream>>>(
      qlm16, kh, vt, LL, SS, w1,
      /*sB=*/(size_t)HH * LL * HD, /*sH=*/(size_t)LL * HD, /*sRow=*/HD);

  // 7) w2^T = (attn2_inv @ w1)^T  -> f16 [bh,64,256]
  gemm_batched_kernel<<<dim3(LL / 64, HD / 64, NBH), 128, 0, stream>>>(
      zc, w1, LL, HD, LL, 0, nullptr, nullptr, 1.f, 0.f, w2t);

  // 8) out = softmax(q @ k_lm^T) @ w2, head-merged to [B,S,512]
  fused_attn_kernel<<<dim3(SS / 16, NBH), 128, 0, stream>>>(
      qh, klm16, w2t, SS, LL, out,
      /*sB=*/(size_t)SS * DD, /*sH=*/(size_t)HD, /*sRow=*/DD);
}
